// InvariantPointAttention_17566416241334
// MI455X (gfx1250) — compile-verified, matrix operand order checked by reading, not run on hardware
//
#include <hip/hip_runtime.h>
#include <math.h>

#define NRES 1024
#define NH   12
#define CC   16
#define PQn  4
#define PVn  8
#define CSd  384
#define CZd  128
#define INFV 100000.0f
#define EPSV 1e-8f
#define ZROWT 4

typedef __attribute__((ext_vector_type(16))) __bf16 v16bf;
typedef __attribute__((ext_vector_type(8)))  float  v8f;
typedef __attribute__((ext_vector_type(4)))  float  f32x4;
typedef __attribute__((ext_vector_type(8)))  unsigned short u16x8;

union BF16x16 { v16bf v; unsigned short u[16]; unsigned d[8]; u16x8 h8[2]; };
union F32x8   { v8f v; float e[8]; };

__device__ __forceinline__ unsigned short f2bfu(float f) {
  unsigned u = __builtin_bit_cast(unsigned, f);
  return (unsigned short)((u + 0x8000u) >> 16);
}
// pack two fp32 -> packed bf16 pair: 2 adds + 1 v_perm_b32
__device__ __forceinline__ unsigned pack_bf2(float f0, float f1) {
  unsigned u0 = __builtin_bit_cast(unsigned, f0) + 0x8000u;
  unsigned u1 = __builtin_bit_cast(unsigned, f1) + 0x8000u;
  return __builtin_amdgcn_perm(u1, u0, 0x07060302u);
}
// K index held by (lane, element h) of a 16-bit fragment, K-step 32
__device__ __forceinline__ int kmap(int lane, int h) {
  return ((h < 8) ? 0 : 16) + ((lane >= 16) ? 8 : 0) + (h & 7);
}
// fragment from fp32 row (global or LDS): 4x b128 loads + 8 packs
__device__ __forceinline__ v16bf load_afrag_f32(const float* rowk0, int lane) {
  int off = (lane & 16) ? 8 : 0;
  f32x4 x0 = *(const f32x4*)(rowk0 + off);
  f32x4 x1 = *(const f32x4*)(rowk0 + off + 4);
  f32x4 x2 = *(const f32x4*)(rowk0 + 16 + off);
  f32x4 x3 = *(const f32x4*)(rowk0 + 16 + off + 4);
  BF16x16 r;
  r.d[0] = pack_bf2(x0.x, x0.y); r.d[1] = pack_bf2(x0.z, x0.w);
  r.d[2] = pack_bf2(x1.x, x1.y); r.d[3] = pack_bf2(x1.z, x1.w);
  r.d[4] = pack_bf2(x2.x, x2.y); r.d[5] = pack_bf2(x2.z, x2.w);
  r.d[6] = pack_bf2(x3.x, x3.y); r.d[7] = pack_bf2(x3.z, x3.w);
  return r.v;
}
// fragment from bf16 row: 2x b128 loads, zero conversion
__device__ __forceinline__ v16bf load_frag_bf16(const unsigned short* rowk0, int lane) {
  int off = (lane & 16) ? 8 : 0;
  BF16x16 r;
  r.h8[0] = *(const u16x8*)(rowk0 + off);
  r.h8[1] = *(const u16x8*)(rowk0 + 16 + off);
  return r.v;
}
// async copy of one contiguous 8KB tile (16 rows x 128 f32) into LDS
__device__ __forceinline__ void async_copy_tile(const float* gsrc, float* ldst, int lane) {
  unsigned lbase = (unsigned)(size_t)ldst + lane * 16u;          // LDS byte offset
  unsigned long long g = (unsigned long long)(size_t)gsrc + (unsigned long long)(lane * 16);
#pragma unroll
  for (int t = 0; t < 16; ++t) {
    unsigned lo = lbase + (unsigned)(t * 512);
    unsigned long long ga = g + (unsigned long long)(t * 512);
    asm volatile("global_load_async_to_lds_b128 %0, %1, off" :: "v"(lo), "v"(ga) : "memory");
  }
}

// ---------------- weight pack: wT[n][k] = bf16(W[k][n]), n padded to mult 16 ----
__global__ void pack_weightT(const float* __restrict__ W, unsigned short* __restrict__ wT,
                             int K, int Nc) {
  int n = blockIdx.x;
  for (int k = threadIdx.x; k < K; k += blockDim.x)
    wT[(size_t)n * K + k] = (n < Nc) ? f2bfu(W[(size_t)k * Nc + n]) : (unsigned short)0;
}

// ---------------- C = A(MxK,f32) @ BT(bf16, transposed KxN) + bias -------------
__global__ void gemm_bf16_wmma(const float* __restrict__ A, const unsigned short* __restrict__ BT,
                               const float* __restrict__ bias, float* __restrict__ Cm,
                               int M, int Nc, int K) {
  int lane = threadIdx.x;
  int Nt = (Nc + 15) / 16;
  int wid = blockIdx.x * blockDim.y + threadIdx.y;
  int mt = wid / Nt, nt = wid % Nt;
  if (mt * 16 >= M) return;
  int arow = mt * 16 + (lane & 15);
  int bcol = nt * 16 + (lane & 15);
  v8f z8 = {};
  F32x8 acc; acc.v = z8;
  const float* Ar = A + (size_t)arow * K;
  const unsigned short* Br = BT + (size_t)bcol * K;
  for (int k0 = 0; k0 < K; k0 += 32) {
    v16bf a = load_afrag_f32(Ar + k0, lane);
    v16bf b = load_frag_bf16(Br + k0, lane);
    acc.v = __builtin_amdgcn_wmma_f32_16x16x32_bf16(false, a, false, b, (short)0, acc.v, false, false);
  }
  int crow0 = mt * 16 + ((lane & 16) ? 8 : 0);
  int ccol  = nt * 16 + (lane & 15);
  if (ccol < Nc) {
    float bb = bias[ccol];
#pragma unroll
    for (int r = 0; r < 8; ++r) {
      int rr = crow0 + r;
      if (rr < M) Cm[(size_t)rr * Nc + ccol] = acc.e[r] + bb;
    }
  }
}

// ---------------- rigid transform of projected points --------------------------
__global__ void point_transform(const float* __restrict__ qp_lin, const float* __restrict__ kvp_lin,
                                const float* __restrict__ rot, const float* __restrict__ trans,
                                float* __restrict__ q_pts, float* __restrict__ k_pts,
                                float* __restrict__ v_pts) {
  int n = blockIdx.x;
  const float* R = rot + n * 9;
  const float* T = trans + n * 3;
  for (int pidx = threadIdx.x; pidx < NH * PQn; pidx += blockDim.x) {
    float p0 = qp_lin[n * (NH * PQn * 3) + 0 * NH * PQn + pidx];
    float p1 = qp_lin[n * (NH * PQn * 3) + 1 * NH * PQn + pidx];
    float p2 = qp_lin[n * (NH * PQn * 3) + 2 * NH * PQn + pidx];
    float x = R[0] * p0 + R[1] * p1 + R[2] * p2 + T[0];
    float y = R[3] * p0 + R[4] * p1 + R[5] * p2 + T[1];
    float z = R[6] * p0 + R[7] * p1 + R[8] * p2 + T[2];
    int h = pidx / PQn, p = pidx % PQn;
    float* d = q_pts + (((size_t)n * NH + h) * PQn + p) * 3;
    d[0] = x; d[1] = y; d[2] = z;
  }
  for (int pidx = threadIdx.x; pidx < NH * 12; pidx += blockDim.x) {
    float p0 = kvp_lin[n * (NH * 36) + 0 * NH * 12 + pidx];
    float p1 = kvp_lin[n * (NH * 36) + 1 * NH * 12 + pidx];
    float p2 = kvp_lin[n * (NH * 36) + 2 * NH * 12 + pidx];
    float x = R[0] * p0 + R[1] * p1 + R[2] * p2 + T[0];
    float y = R[3] * p0 + R[4] * p1 + R[5] * p2 + T[1];
    float z = R[6] * p0 + R[7] * p1 + R[8] * p2 + T[2];
    int h = pidx / 12, p = pidx % 12;
    float* d = (p < PQn)
      ? k_pts + (((size_t)n * NH + h) * PQn + p) * 3
      : v_pts + (((size_t)n * NH + h) * PVn + (p - PQn)) * 3;
    d[0] = x; d[1] = y; d[2] = z;
  }
}

__global__ void point_norms(const float* __restrict__ q_pts, const float* __restrict__ k_pts,
                            float* __restrict__ qn, float* __restrict__ kn) {
  int idx = blockIdx.x * blockDim.x + threadIdx.x;
  if (idx >= NRES * NH) return;
  float sq = 0.f, sk = 0.f;
  for (int t = 0; t < 12; ++t) {
    float a = q_pts[(size_t)idx * 12 + t]; sq += a * a;
    float b = k_pts[(size_t)idx * 12 + t]; sk += b * b;
  }
  qn[idx] = sq; kn[idx] = sk;
}

// ------- pack logit operands: [h][n][32] bf16, scales folded into q side --------
__global__ void pack_qk(const float* __restrict__ q_lin, const float* __restrict__ kv_lin,
                        const float* __restrict__ q_pts, const float* __restrict__ k_pts,
                        const float* __restrict__ head_weights,
                        unsigned short* __restrict__ qpk, unsigned short* __restrict__ kpk) {
  int n = blockIdx.x;
  float sQK = sqrtf(1.0f / (3.0f * CC));
  for (int idx = threadIdx.x; idx < NH * 32; idx += blockDim.x) {
    int h = idx >> 5, k = idx & 31;
    float qv = 0.f, kv = 0.f;
    if (k < CC) {
      qv = q_lin[(size_t)n * (NH * CC) + h * CC + k] * sQK;
      kv = kv_lin[(size_t)n * (2 * NH * CC) + h * (2 * CC) + k];
    } else if (k < CC + 12) {
      float hw = log1pf(expf(head_weights[h])) * sqrtf(1.0f / 54.0f);
      int pd = k - CC;
      qv = q_pts[((size_t)n * NH + h) * 12 + pd] * hw;
      kv = k_pts[((size_t)n * NH + h) * 12 + pd];
    }
    qpk[((size_t)h * NRES + n) * 32 + k] = f2bfu(qv);
    kpk[((size_t)h * NRES + n) * 32 + k] = f2bfu(kv);
  }
}

// ------- pack V operands transposed bf16: vT[h][c][j], vpT[h][pd(0..31)][j] -----
__global__ void pack_v(const float* __restrict__ kv_lin, const float* __restrict__ v_pts,
                       unsigned short* __restrict__ vT, unsigned short* __restrict__ vpT) {
  int j = blockIdx.x;
  for (int idx = threadIdx.x; idx < NH * CC; idx += blockDim.x) {
    int h = idx >> 4, c = idx & 15;
    vT[((size_t)h * CC + c) * NRES + j] =
        f2bfu(kv_lin[(size_t)j * (2 * NH * CC) + h * (2 * CC) + CC + c]);
  }
  for (int idx = threadIdx.x; idx < NH * 32; idx += blockDim.x) {
    int h = idx >> 5, pd = idx & 31;
    float v = (pd < 24) ? v_pts[((size_t)j * NH + h) * 24 + pd] : 0.f;
    vpT[((size_t)h * 32 + pd) * NRES + j] = f2bfu(v);
  }
}

// ---- fused z projection: async double-buffered LDS staging of z, read ONCE -----
// bpT (f32, [h][i][j]) + pzT (bf16, transposed [i][c][j])
__global__ void __launch_bounds__(64) zproj_wmma(
    const float* __restrict__ z,
    const float* __restrict__ w_b, const float* __restrict__ b_b,
    const float* __restrict__ w_dz, const float* __restrict__ b_dz,
    float* __restrict__ bpT, unsigned short* __restrict__ pzT) {
  __shared__ float lbuf[2][2][16 * CZd];   // [wave][buf][16x128 tile] = 32 KB
  int lane = threadIdx.x;
  long wv = (long)blockIdx.x * blockDim.y + threadIdx.y;
  long tile0 = wv * ZROWT;
  int bcol = lane & 15;
  // Preload all 12 weight B-fragments (held in registers for the whole wave)
  BF16x16 bf[4][3];
#pragma unroll
  for (int ks = 0; ks < 4; ++ks)
#pragma unroll
    for (int t = 0; t < 3; ++t)
#pragma unroll
      for (int h = 0; h < 16; ++h) {
        int kk = ks * 32 + kmap(lane, h);
        int col = t * 16 + bcol;
        float wvv = (col < NH) ? w_b[kk * NH + col]
                  : ((col < NH + 32) ? w_dz[kk * 32 + (col - NH)] : 0.f);
        bf[ks][t].u[h] = f2bfu(wvv);
      }
  float (*my)[16 * CZd] = lbuf[threadIdx.y];
  // prologue: stage tile 0
  async_copy_tile(z + tile0 * 16 * CZd, my[0], lane);
  for (int rt = 0; rt < ZROWT; ++rt) {
    if (rt + 1 < ZROWT) {
      async_copy_tile(z + (tile0 + rt + 1) * 16 * CZd, my[(rt + 1) & 1], lane);
      asm volatile("s_wait_asynccnt 0x10" ::: "memory");   // tile rt complete
    } else {
      asm volatile("s_wait_asynccnt 0x0" ::: "memory");
    }
    const float* lrow = my[rt & 1] + (lane & 15) * CZd;    // ds_load_b128 reads
    v8f z8 = {};
    F32x8 a0, a1, a2; a0.v = z8; a1.v = z8; a2.v = z8;
#pragma unroll
    for (int ks = 0; ks < 4; ++ks) {
      v16bf a = load_afrag_f32(lrow + ks * 32, lane);
      a0.v = __builtin_amdgcn_wmma_f32_16x16x32_bf16(false, a, false, bf[ks][0].v, (short)0, a0.v, false, false);
      a1.v = __builtin_amdgcn_wmma_f32_16x16x32_bf16(false, a, false, bf[ks][1].v, (short)0, a1.v, false, false);
      a2.v = __builtin_amdgcn_wmma_f32_16x16x32_bf16(false, a, false, bf[ks][2].v, (short)0, a2.v, false, false);
    }
    long row16 = (tile0 + rt) * 16;
    long crow0 = row16 + ((lane & 16) ? 8 : 0);
    long i_idx = crow0 >> 10;
    int  j0    = (int)(crow0 & 1023);
    if (bcol < NH) {          // acc0 cols 0..11 -> bpT[h][flat], 2x b128 stores
      float bb = b_b[bcol];
      f32x4 s0 = { a0.e[0] + bb, a0.e[1] + bb, a0.e[2] + bb, a0.e[3] + bb };
      f32x4 s1 = { a0.e[4] + bb, a0.e[5] + bb, a0.e[6] + bb, a0.e[7] + bb };
      float* dst = bpT + (((size_t)bcol) << 20) + crow0;
      *(f32x4*)dst = s0;
      *(f32x4*)(dst + 4) = s1;
    } else {                  // acc0 cols 12..15 -> pz c=0..3
      int c = bcol - NH;
      float bb = b_dz[c];
      unsigned short* bptr = pzT + ((size_t)i_idx * 32 + c) * NRES + j0;
#pragma unroll
      for (int r = 0; r < 8; r += 2)
        *(unsigned*)(bptr + r) = pack_bf2(a0.e[r] + bb, a0.e[r + 1] + bb);
    }
    {                         // acc1 cols 16..31 -> pz c=4..19
      int c = bcol + 4;
      float bb = b_dz[c];
      unsigned short* bptr = pzT + ((size_t)i_idx * 32 + c) * NRES + j0;
#pragma unroll
      for (int r = 0; r < 8; r += 2)
        *(unsigned*)(bptr + r) = pack_bf2(a1.e[r] + bb, a1.e[r + 1] + bb);
    }
    if (bcol < NH) {          // acc2 cols 32..43 -> pz c=20..31
      int c = bcol + 20;
      float bb = b_dz[c];
      unsigned short* bptr = pzT + ((size_t)i_idx * 32 + c) * NRES + j0;
#pragma unroll
      for (int r = 0; r < 8; r += 2)
        *(unsigned*)(bptr + r) = pack_bf2(a2.e[r] + bb, a2.e[r + 1] + bb);
    }
  }
}

// ------------- logits: one K=32 WMMA (QK dot + point dot), fp32 epilogue --------
__global__ void logits_wmma(const unsigned short* __restrict__ qpk,
                            const unsigned short* __restrict__ kpk,
                            const float* __restrict__ qn, const float* __restrict__ kn,
                            const float* __restrict__ bpT, const float* __restrict__ mask,
                            const float* __restrict__ head_weights,
                            float* __restrict__ logits) {
  int lane = threadIdx.x;
  int h = blockIdx.z, it = blockIdx.x;
  int jt = blockIdx.y * blockDim.y + threadIdx.y;
  int qrow = it * 16 + (lane & 15);
  int kcol = jt * 16 + (lane & 15);
  v16bf a = load_frag_bf16(qpk + ((size_t)h * NRES + qrow) * 32, lane);
  v16bf b = load_frag_bf16(kpk + ((size_t)h * NRES + kcol) * 32, lane);
  v8f z8 = {};
  F32x8 acc; acc.v = z8;
  acc.v = __builtin_amdgcn_wmma_f32_16x16x32_bf16(false, a, false, b, (short)0, acc.v, false, false);
  int crow0 = it * 16 + ((lane & 16) ? 8 : 0);
  int ccol  = jt * 16 + (lane & 15);
  float hw  = log1pf(expf(head_weights[h])) * sqrtf(1.0f / 54.0f);
  float sq3 = sqrtf(1.0f / 3.0f);
  float mj  = mask[ccol];
  float knj = kn[ccol * NH + h];
  const float* bph = bpT + (((size_t)h) << 20) + ccol;  // [h][i][j], lanes span 64B rows
#pragma unroll
  for (int r = 0; r < 8; ++r) {
    int rr = crow0 + r;
    float v = acc.e[r]
            + sq3 * bph[(size_t)rr * NRES]
            - 0.5f * hw * (qn[rr * NH + h] + knj)
            + INFV * (mask[rr] * mj - 1.0f);
    logits[((size_t)h * NRES + rr) * NRES + ccol] = v;
  }
}

// ---------------- rowwise softmax, bf16 prob output ----------------------------
__global__ void softmax_rows(float* __restrict__ logits, unsigned short* __restrict__ probs) {
  size_t row = blockIdx.x;
  float* p = logits + row * NRES;
  unsigned short* o = probs + row * NRES;
  __shared__ float red[256];
  int tid = threadIdx.x;
  float m = -1e30f;
  for (int j = tid; j < NRES; j += 256) m = fmaxf(m, p[j]);
  red[tid] = m; __syncthreads();
  for (int s = 128; s > 0; s >>= 1) { if (tid < s) red[tid] = fmaxf(red[tid], red[tid + s]); __syncthreads(); }
  m = red[0]; __syncthreads();
  float sum = 0.f;
  for (int j = tid; j < NRES; j += 256) { float e = expf(p[j] - m); p[j] = e; sum += e; }
  red[tid] = sum; __syncthreads();
  for (int s = 128; s > 0; s >>= 1) { if (tid < s) red[tid] += red[tid + s]; __syncthreads(); }
  float inv = 1.0f / red[0];
  for (int j = tid * 2; j < NRES; j += 512)
    *(unsigned*)(o + j) = pack_bf2(p[j] * inv, p[j + 1] * inv);
}

// ---------------- o = a@v, o_pt = a@v_pts: all-b128 operands --------------------
__global__ void attn_out_wmma(const unsigned short* __restrict__ amat,
                              const unsigned short* __restrict__ vT,
                              const unsigned short* __restrict__ vpT,
                              float* __restrict__ o, float* __restrict__ o_pt_raw) {
  int lane = threadIdx.x;
  int wid = blockIdx.x * blockDim.y + threadIdx.y;
  int h = wid % NH, it = wid / NH;
  if (it >= NRES / 16) return;
  v8f z8 = {};
  F32x8 accO, accP0, accP1; accO.v = z8; accP0.v = z8; accP1.v = z8;
  int arow = it * 16 + (lane & 15);
  int bcol = lane & 15;
  const unsigned short* ap  = amat + ((size_t)h * NRES + arow) * NRES;
  const unsigned short* vr  = vT  + ((size_t)h * CC + bcol) * NRES;
  const unsigned short* vp0 = vpT + ((size_t)h * 32 + bcol) * NRES;
  const unsigned short* vp1 = vpT + ((size_t)h * 32 + 16 + bcol) * NRES;
  for (int k0 = 0; k0 < NRES; k0 += 32) {
    v16bf a  = load_frag_bf16(ap  + k0, lane);
    v16bf bv = load_frag_bf16(vr  + k0, lane);
    v16bf b0 = load_frag_bf16(vp0 + k0, lane);
    v16bf b1 = load_frag_bf16(vp1 + k0, lane);
    accO.v  = __builtin_amdgcn_wmma_f32_16x16x32_bf16(false, a, false, bv, (short)0, accO.v,  false, false);
    accP0.v = __builtin_amdgcn_wmma_f32_16x16x32_bf16(false, a, false, b0, (short)0, accP0.v, false, false);
    accP1.v = __builtin_amdgcn_wmma_f32_16x16x32_bf16(false, a, false, b1, (short)0, accP1.v, false, false);
  }
  int crow0 = it * 16 + ((lane & 16) ? 8 : 0);
#pragma unroll
  for (int r = 0; r < 8; ++r) {
    int rr = crow0 + r;
    o[(size_t)rr * (NH * CC) + h * CC + bcol] = accO.e[r];
    o_pt_raw[((size_t)rr * NH + h) * 24 + bcol] = accP0.e[r];
    if (16 + bcol < 24) o_pt_raw[((size_t)rr * NH + h) * 24 + 16 + bcol] = accP1.e[r];
  }
}

// ---------------- o_pair[i] = a[:,i,:] @ pair_z[i]: all-b128 operands -----------
__global__ void opair_wmma(const unsigned short* __restrict__ amat,
                           const unsigned short* __restrict__ pzT,
                           float* __restrict__ o_pair) {
  int lane = threadIdx.x;
  int i = blockIdx.x * blockDim.y + threadIdx.y;
  if (i >= NRES) return;
  int hrow = lane & 15;
  v8f z8 = {};
  F32x8 acc0, acc1; acc0.v = z8; acc1.v = z8;
  const unsigned short* ar = amat + ((size_t)hrow * NRES + i) * NRES;
  const unsigned short* p0 = pzT + ((size_t)i * 32 + (lane & 15)) * NRES;
  const unsigned short* p1 = pzT + ((size_t)i * 32 + 16 + (lane & 15)) * NRES;
  u16x8 z16 = {};
  for (int k0 = 0; k0 < NRES; k0 += 32) {
    BF16x16 a;
    if (hrow < NH) a.v = load_frag_bf16(ar + k0, lane);
    else { a.h8[0] = z16; a.h8[1] = z16; }
    v16bf b0 = load_frag_bf16(p0 + k0, lane);
    v16bf b1 = load_frag_bf16(p1 + k0, lane);
    acc0.v = __builtin_amdgcn_wmma_f32_16x16x32_bf16(false, a.v, false, b0, (short)0, acc0.v, false, false);
    acc1.v = __builtin_amdgcn_wmma_f32_16x16x32_bf16(false, a.v, false, b1, (short)0, acc1.v, false, false);
  }
  int hr0 = (lane & 16) ? 8 : 0;
#pragma unroll
  for (int r = 0; r < 8; ++r) {
    int hh = hr0 + r;
    if (hh < NH) {
      o_pair[((size_t)i * NH + hh) * 32 + (lane & 15)]      = acc0.e[r];
      o_pair[((size_t)i * NH + hh) * 32 + 16 + (lane & 15)] = acc1.e[r];
    }
  }
}

// ---------------- epilogue: rotate back, norms, concat --------------------------
__global__ void build_cat(const float* __restrict__ o, const float* __restrict__ o_pt_raw,
                          const float* __restrict__ o_pair, const float* __restrict__ rot,
                          const float* __restrict__ trans, float* __restrict__ cat) {
  int n = blockIdx.x, tid = threadIdx.x;
  const float* R = rot + n * 9;
  const float* T = trans + n * 3;
  float* crow = cat + (size_t)n * 960;
  for (int idx = tid; idx < NH * CC; idx += blockDim.x) crow[idx] = o[(size_t)n * NH * CC + idx];
  for (int idx = tid; idx < NH * 32; idx += blockDim.x) crow[576 + idx] = o_pair[((size_t)n * NH) * 32 + idx];
  for (int idx = tid; idx < NH * PVn; idx += blockDim.x) {
    int h = idx / PVn, p = idx % PVn;
    const float* s = o_pt_raw + ((size_t)n * NH + h) * 24 + p * 3;
    float gx = s[0] - T[0], gy = s[1] - T[1], gz = s[2] - T[2];
    float lx = R[0] * gx + R[3] * gy + R[6] * gz;
    float ly = R[1] * gx + R[4] * gy + R[7] * gz;
    float lz = R[2] * gx + R[5] * gy + R[8] * gz;
    crow[192 + idx] = lx;
    crow[288 + idx] = ly;
    crow[384 + idx] = lz;
    crow[480 + idx] = sqrtf(lx * lx + ly * ly + lz * lz + EPSV);
  }
}

extern "C" void kernel_launch(void* const* d_in, const int* in_sizes, int n_in,
                              void* d_out, int out_size, void* d_ws, size_t ws_size,
                              hipStream_t stream) {
  (void)in_sizes; (void)n_in; (void)out_size; (void)ws_size;
  const float* s     = (const float*)d_in[0];
  const float* z     = (const float*)d_in[1];
  const float* rot   = (const float*)d_in[2];
  const float* trans = (const float*)d_in[3];
  const float* mask  = (const float*)d_in[4];
  const float* w_q   = (const float*)d_in[5];
  const float* b_q   = (const float*)d_in[6];
  const float* w_kv  = (const float*)d_in[7];
  const float* b_kv  = (const float*)d_in[8];
  const float* w_qp  = (const float*)d_in[9];
  const float* b_qp  = (const float*)d_in[10];
  const float* w_kvp = (const float*)d_in[11];
  const float* b_kvp = (const float*)d_in[12];
  const float* w_b   = (const float*)d_in[13];
  const float* b_b   = (const float*)d_in[14];
  const float* w_dz  = (const float*)d_in[15];
  const float* b_dz  = (const float*)d_in[16];
  const float* hwts  = (const float*)d_in[17];
  const float* w_out = (const float*)d_in[18];
  const float* b_out = (const float*)d_in[19];
  float* out = (float*)d_out;

  char* wp = (char*)d_ws;
  auto alloc = [&](size_t bytes) { void* r = (void*)wp; wp += (bytes + 255) & ~(size_t)255; return r; };
  float* q_lin   = (float*)alloc((size_t)NRES * 192 * 4);
  float* kv_lin  = (float*)alloc((size_t)NRES * 384 * 4);
  float* qp_lin  = (float*)alloc((size_t)NRES * 144 * 4);
  float* kvp_lin = (float*)alloc((size_t)NRES * 432 * 4);
  float* q_pts   = (float*)alloc((size_t)NRES * NH * 12 * 4);
  float* k_pts   = (float*)alloc((size_t)NRES * NH * 12 * 4);
  float* v_pts   = (float*)alloc((size_t)NRES * NH * 24 * 4);
  float* qn      = (float*)alloc((size_t)NRES * NH * 4);
  float* kn      = (float*)alloc((size_t)NRES * NH * 4);
  float* o_buf   = (float*)alloc((size_t)NRES * 192 * 4);
  float* o_ptr   = (float*)alloc((size_t)NRES * NH * 24 * 4);
  float* o_pair  = (float*)alloc((size_t)NRES * NH * 32 * 4);
  float* catb    = (float*)alloc((size_t)NRES * 960 * 4);
  unsigned short* wqT   = (unsigned short*)alloc((size_t)192 * CSd * 2);
  unsigned short* wkvT  = (unsigned short*)alloc((size_t)384 * CSd * 2);
  unsigned short* wqpT  = (unsigned short*)alloc((size_t)144 * CSd * 2);
  unsigned short* wkvpT = (unsigned short*)alloc((size_t)432 * CSd * 2);
  unsigned short* woutT = (unsigned short*)alloc((size_t)CSd * 960 * 2);
  unsigned short* qpk   = (unsigned short*)alloc((size_t)NH * NRES * 32 * 2);
  unsigned short* kpk   = (unsigned short*)alloc((size_t)NH * NRES * 32 * 2);
  unsigned short* vT    = (unsigned short*)alloc((size_t)NH * CC * NRES * 2);
  unsigned short* vpT   = (unsigned short*)alloc((size_t)NH * 32 * NRES * 2);
  float*          bpT   = (float*)alloc((size_t)NH * NRES * NRES * 4);
  float*          lgt   = (float*)alloc((size_t)NH * NRES * NRES * 4);
  unsigned short* pzT   = (unsigned short*)alloc((size_t)NRES * 32 * NRES * 2);
  unsigned short* amat  = (unsigned short*)alloc((size_t)NH * NRES * NRES * 2);

  dim3 blk(32, 8);
  auto gblocks = [](int M, int Nc) {
    int Mt = (M + 15) / 16, Nt = (Nc + 15) / 16;
    return dim3((unsigned)((Mt * (long)Nt + 7) / 8));
  };

  // 0) pack weights transposed bf16 (one-time, tiny)
  pack_weightT<<<192, 128, 0, stream>>>(w_q,   wqT,   CSd, 192);
  pack_weightT<<<384, 128, 0, stream>>>(w_kv,  wkvT,  CSd, 384);
  pack_weightT<<<144, 128, 0, stream>>>(w_qp,  wqpT,  CSd, 144);
  pack_weightT<<<432, 128, 0, stream>>>(w_kvp, wkvpT, CSd, 432);
  pack_weightT<<<384, 128, 0, stream>>>(w_out, woutT, 960, CSd);

  // 1) s-projections
  gemm_bf16_wmma<<<gblocks(NRES, 192), blk, 0, stream>>>(s, wqT,   b_q,   q_lin,   NRES, 192, CSd);
  gemm_bf16_wmma<<<gblocks(NRES, 384), blk, 0, stream>>>(s, wkvT,  b_kv,  kv_lin,  NRES, 384, CSd);
  gemm_bf16_wmma<<<gblocks(NRES, 144), blk, 0, stream>>>(s, wqpT,  b_qp,  qp_lin,  NRES, 144, CSd);
  gemm_bf16_wmma<<<gblocks(NRES, 432), blk, 0, stream>>>(s, wkvpT, b_kvp, kvp_lin, NRES, 432, CSd);

  // 2) transforms + operand packing
  point_transform<<<NRES, 64, 0, stream>>>(qp_lin, kvp_lin, rot, trans, q_pts, k_pts, v_pts);
  point_norms<<<(NRES * NH + 255) / 256, 256, 0, stream>>>(q_pts, k_pts, qn, kn);
  pack_qk<<<NRES, 128, 0, stream>>>(q_lin, kv_lin, q_pts, k_pts, hwts, qpk, kpk);
  pack_v<<<NRES, 128, 0, stream>>>(kv_lin, v_pts, vT, vpT);

  // 3) fused z projection: async double-buffered LDS staging, z streamed once
  zproj_wmma<<<(NRES / 16) * (NRES / 16) / (ZROWT * 2), dim3(32, 2), 0, stream>>>(
      z, w_b, b_b, w_dz, b_dz, bpT, pzT);

  // 4) logits + softmax -> bf16 probs
  logits_wmma<<<dim3(NRES / 16, NRES / 16 / 8, NH), blk, 0, stream>>>(
      qpk, kpk, qn, kn, bpT, mask, hwts, lgt);
  softmax_rows<<<NH * NRES, 256, 0, stream>>>(lgt, amat);

  // 5) attention outputs (pure b128 + WMMA)
  attn_out_wmma<<<(NRES / 16) * NH / 8, blk, 0, stream>>>(amat, vT, vpT, o_buf, o_ptr);
  opair_wmma<<<NRES / 8, blk, 0, stream>>>(amat, pzT, o_pair);

  // 6) epilogue + final projection
  build_cat<<<NRES, 128, 0, stream>>>(o_buf, o_ptr, o_pair, rot, trans, catb);
  gemm_bf16_wmma<<<gblocks(NRES, CSd), blk, 0, stream>>>(catb, woutT, b_out, out, NRES, CSd, 960);
}
